// Experimental_RouteNet_Model_89644557402401
// MI455X (gfx1250) — matrix operand order repaired and checked by reading, not compiled, not run on hardware
//
#include <hip/hip_runtime.h>
#include <hip/hip_bf16.h>

// ---------------- problem constants ----------------
#define PN   100000   // paths
#define LN   5        // path length
#define NLN  10000    // links
#define DN   32       // state dim
#define RUN  256      // readout units
#define TN   8        // MP iterations

typedef _Float16 v16h __attribute__((ext_vector_type(16)));
typedef float    v8f  __attribute__((ext_vector_type(8)));

__device__ __forceinline__ void lds_fence() {
  // CDNA5 split wait counter for LDS ops (wave-internal producer/consumer)
  asm volatile("s_wait_dscnt 0" ::: "memory");
}

__device__ __forceinline__ float fsigmoid(float x) {
  return 1.0f / (1.0f + __expf(-x));
}
__device__ __forceinline__ float ftanh(float x) {
  float e = __expf(2.0f * x);
  return (e - 1.0f) / (e + 1.0f);
}
__device__ __forceinline__ float fselu(float x) {
  const float scale = 1.0507009873554805f, alpha = 1.6732632423543772f;
  return scale * (x > 0.0f ? x : alpha * (__expf(x) - 1.0f));
}

// 16-bit A-fragment K index for linear half i (0..15):
// lanes 0-15 hold K {0..7,16..23}; lanes 16-31 hold K {8..15,24..31}
__device__ __forceinline__ int a_k(int i, int lane) {
  return i + ((i >= 8) ? 8 : 0) + ((lane >= 16) ? 8 : 0);
}

// B fragment: row-major f16 matrix, K-chunk [k0,k0+32), N tile [n0,n0+16).
// lane holds column n0+(lane&15); lanes 0-15 cover K k0+{0..7,16..23}, lanes 16-31 K k0+{8..15,24..31}.
__device__ __forceinline__ v16h load_b_frag(const _Float16* __restrict__ B, int ldb,
                                            int k0, int n0, int lane) {
  v16h f;
  const int n  = n0 + (lane & 15);
  const int kb = k0 + ((lane >> 4) << 3);
#pragma unroll
  for (int i = 0; i < 16; ++i) f[i] = B[(kb + i + ((i >= 8) ? 8 : 0)) * ldb + n];
  return f;
}

// ---------------- init & prep kernels ----------------
__global__ __launch_bounds__(256) void init_states_kernel(
    const float* __restrict__ cap, const float* __restrict__ pol,
    const float* __restrict__ twt, const float* __restrict__ bw,
    const float* __restrict__ tos, const float* __restrict__ pk,
    const float* __restrict__ avg,
    float* __restrict__ link_state, float* __restrict__ path_state) {
  int i = blockIdx.x * blockDim.x + threadIdx.x;
  if (i < NLN * DN) {
    int l = i >> 5, c = i & 31;
    link_state[i] = (c == 0) ? cap[l] : (c == 1) ? pol[l] : (c == 2) ? twt[l] : 0.0f;
  }
  if (i < PN * DN) {
    int p = i >> 5, c = i & 31;
    path_state[i] = (c == 0) ? bw[p] : (c == 1) ? tos[p] : (c == 2) ? pk[p]
                  : (c == 3) ? avg[p] : 0.0f;
  }
}

// f32 -> f16 weight conversion into workspace
// layout: Wp[3072] Up[3072] Wl[3072] Ul[3072] R1[8192] R2[65536]  (total 86016 halves)
__global__ __launch_bounds__(256) void cvt_weights_kernel(
    const float* __restrict__ Wp, const float* __restrict__ Up,
    const float* __restrict__ Wl, const float* __restrict__ Ul,
    const float* __restrict__ R1, const float* __restrict__ R2,
    _Float16* __restrict__ w16) {
  int i = blockIdx.x * blockDim.x + threadIdx.x;
  if (i >= 86016) return;
  float v;
  if      (i <  3072) v = Wp[i];
  else if (i <  6144) v = Up[i - 3072];
  else if (i <  9216) v = Wl[i - 6144];
  else if (i < 12288) v = Ul[i - 9216];
  else if (i < 20480) v = R1[i - 12288];
  else                v = R2[i - 20480];
  w16[i] = (_Float16)v;
}

__global__ __launch_bounds__(256) void zero_kernel(float* __restrict__ p, int n) {
  int i = blockIdx.x * blockDim.x + threadIdx.x;
  if (i < n) p[i] = 0.0f;
}

// ---------------- path GRU scan + scatter (per MP iteration) ----------------
// one wave == 16 paths; 5 GRU steps; x_t gathered from link_state; outputs
// scatter-added into m_agg[link] (segment_sum) at every step.
__global__ __launch_bounds__(128) void path_mp_kernel(
    const int*   __restrict__ links,
    const float* __restrict__ link_state,
    float*       __restrict__ path_state,
    float*       __restrict__ m_agg,
    const _Float16* __restrict__ Wp16,
    const _Float16* __restrict__ Up16,
    const float* __restrict__ bp) {
  __shared__ float hrow[4][16][DN + 1];
  const int lane = threadIdx.x & 31;
  const int wid  = threadIdx.x >> 5;
  const int tile = blockIdx.x * 4 + wid;
  if (tile * 16 >= PN) return;           // wave-uniform -> EXEC all-ones for WMMA
  const int p0 = tile * 16;
  float (*h)[DN + 1] = hrow[wid];

  const int mrow = lane & 15;            // A-fragment row owned by this lane
  const int crow = (lane >> 4) << 3;     // C-fragment row base
  const int ccol = lane & 15;            // C-fragment column (within N tile)

  // weight fragments resident in VGPRs (6 N-tiles of 16 for the 96-wide gates)
  v16h wf[6], uf[6];
  float bx[6], bh[6];
#pragma unroll
  for (int j = 0; j < 6; ++j) {
    wf[j] = load_b_frag(Wp16, 96, 0, j * 16, lane);
    uf[j] = load_b_frag(Up16, 96, 0, j * 16, lane);
    bx[j] = bp[j * 16 + ccol];           // input bias  b[0]
    bh[j] = bp[96 + j * 16 + ccol];      // recurrent bias b[1]
  }

  // stage h tile row-major in LDS (source for recurrent A-fragments)
  for (int r = 0; r < 16; ++r) h[r][lane] = path_state[(p0 + r) * DN + lane];
  lds_fence();

  for (int t = 0; t < LN; ++t) {
    // x_t A-fragment gathered straight from link_state rows (seqs/paths are regular)
    const int lk = links[(p0 + mrow) * LN + t];
    const float* __restrict__ xr = link_state + lk * DN;
    v16h xa, ha;
#pragma unroll
    for (int i = 0; i < 16; ++i) {
      int k = a_k(i, lane);
      xa[i] = (_Float16)xr[k];
      ha[i] = (_Float16)h[mrow][k];
    }
    v8f gx[6], gh[6];
#pragma unroll
    for (int j = 0; j < 6; ++j) {
      v8f z0 = {};
      gx[j] = __builtin_amdgcn_wmma_f32_16x16x32_f16(false, xa, false, wf[j],
                                                     (short)0, z0, false, false);
      gh[j] = __builtin_amdgcn_wmma_f32_16x16x32_f16(false, ha, false, uf[j],
                                                     (short)0, z0, false, false);
    }
    // links for the 8 C-rows this lane owns (for the scatter)
    int lk8[8];
#pragma unroll
    for (int r = 0; r < 8; ++r) lk8[r] = links[(p0 + crow + r) * LN + t];

    lds_fence();
#pragma unroll
    for (int j = 0; j < 2; ++j) {        // N-tiles 0,1 cover state cols 0..31
#pragma unroll
      for (int r = 0; r < 8; ++r) {
        float zz = fsigmoid(gx[j][r] + bx[j] + gh[j][r] + bh[j]);
        float rr = fsigmoid(gx[j + 2][r] + bx[j + 2] + gh[j + 2][r] + bh[j + 2]);
        float nn = ftanh(gx[j + 4][r] + bx[j + 4] + rr * (gh[j + 4][r] + bh[j + 4]));
        float hv = h[crow + r][j * 16 + ccol];
        float hn = zz * hv + (1.0f - zz) * nn;        // mask is all-true (lens == L)
        h[crow + r][j * 16 + ccol] = hn;
        atomicAdd(&m_agg[lk8[r] * DN + j * 16 + ccol], hn);  // segment_sum over links
      }
    }
  }
  lds_fence();
  for (int r = 0; r < 16; ++r) path_state[(p0 + r) * DN + lane] = h[r][lane];
}

// ---------------- link GRU (one step per MP iteration) ----------------
__global__ __launch_bounds__(128) void link_mp_kernel(
    const float* __restrict__ m_agg,
    float*       __restrict__ link_state,
    const _Float16* __restrict__ Wl16,
    const _Float16* __restrict__ Ul16,
    const float* __restrict__ bl) {
  const int lane = threadIdx.x & 31;
  const int wid  = threadIdx.x >> 5;
  const int tile = blockIdx.x * 4 + wid;
  if (tile * 16 >= NLN) return;
  const int l0 = tile * 16;

  const int mrow = lane & 15;
  const int crow = (lane >> 4) << 3;
  const int ccol = lane & 15;

  v16h xa, ha;
#pragma unroll
  for (int i = 0; i < 16; ++i) {
    int k = a_k(i, lane);
    xa[i] = (_Float16)m_agg[(l0 + mrow) * DN + k];
    ha[i] = (_Float16)link_state[(l0 + mrow) * DN + k];
  }
  v8f gx[6], gh[6];
#pragma unroll
  for (int j = 0; j < 6; ++j) {
    v8f z0 = {};
    v16h wf = load_b_frag(Wl16, 96, 0, j * 16, lane);
    v16h uf = load_b_frag(Ul16, 96, 0, j * 16, lane);
    gx[j] = __builtin_amdgcn_wmma_f32_16x16x32_f16(false, xa, false, wf,
                                                   (short)0, z0, false, false);
    gh[j] = __builtin_amdgcn_wmma_f32_16x16x32_f16(false, ha, false, uf,
                                                   (short)0, z0, false, false);
  }
#pragma unroll
  for (int j = 0; j < 2; ++j) {
#pragma unroll
    for (int r = 0; r < 8; ++r) {
      float bxj = bl[j * 16 + ccol],       bhj = bl[96 + j * 16 + ccol];
      float bxr = bl[(j + 2) * 16 + ccol], bhr = bl[96 + (j + 2) * 16 + ccol];
      float bxn = bl[(j + 4) * 16 + ccol], bhn = bl[96 + (j + 4) * 16 + ccol];
      float zz = fsigmoid(gx[j][r] + bxj + gh[j][r] + bhj);
      float rr = fsigmoid(gx[j + 2][r] + bxr + gh[j + 2][r] + bhr);
      float nn = ftanh(gx[j + 4][r] + bxn + rr * (gh[j + 4][r] + bhn));
      float hv = link_state[(l0 + crow + r) * DN + j * 16 + ccol];
      link_state[(l0 + crow + r) * DN + j * 16 + ccol] = zz * hv + (1.0f - zz) * nn;
    }
  }
}

// ---------------- readout MLP: 32 -> 256 (selu) -> 256 (selu) -> 1 ----------------
__global__ __launch_bounds__(64) void readout_kernel(
    const float* __restrict__ path_state,
    const _Float16* __restrict__ R1h, const float* __restrict__ Rb1,
    const _Float16* __restrict__ R2h, const float* __restrict__ Rb2,
    const float* __restrict__ R3,     const float* __restrict__ Rb3,
    float* __restrict__ out) {
  __shared__ _Float16 h1s[2][16][RUN + 8];
  const int lane = threadIdx.x & 31;
  const int wid  = threadIdx.x >> 5;
  const int tile = blockIdx.x * 2 + wid;
  const int p0   = tile * 16;               // PN/16 divides the grid exactly

  const int mrow = lane & 15;
  const int crow = (lane >> 4) << 3;
  const int ccol = lane & 15;

  v16h xa;
#pragma unroll
  for (int i = 0; i < 16; ++i)
    xa[i] = (_Float16)path_state[(p0 + mrow) * DN + a_k(i, lane)];

  // layer 1: [16,32] x [32,256]
  for (int nt = 0; nt < 16; ++nt) {
    v8f acc = {};
    v16h bf = load_b_frag(R1h, RUN, 0, nt * 16, lane);
    acc = __builtin_amdgcn_wmma_f32_16x16x32_f16(false, xa, false, bf,
                                                 (short)0, acc, false, false);
    float bias = Rb1[nt * 16 + ccol];
#pragma unroll
    for (int r = 0; r < 8; ++r)
      h1s[wid][crow + r][nt * 16 + ccol] = (_Float16)fselu(acc[r] + bias);
  }
  lds_fence();

  // layer 2 + layer 3 fused: [16,256] x [256,256] -> selu -> dot with R3
  float part[8] = {0.f, 0.f, 0.f, 0.f, 0.f, 0.f, 0.f, 0.f};
  for (int nt = 0; nt < 16; ++nt) {
    v8f acc = {};
    for (int kc = 0; kc < 8; ++kc) {
      v16h a2;
#pragma unroll
      for (int i = 0; i < 16; ++i)
        a2[i] = h1s[wid][mrow][kc * 32 + a_k(i, lane)];
      v16h bf = load_b_frag(R2h, RUN, kc * 32, nt * 16, lane);
      acc = __builtin_amdgcn_wmma_f32_16x16x32_f16(false, a2, false, bf,
                                                   (short)0, acc, false, false);
    }
    float bias = Rb2[nt * 16 + ccol];
    float w3   = R3[nt * 16 + ccol];
#pragma unroll
    for (int r = 0; r < 8; ++r) part[r] += fselu(acc[r] + bias) * w3;
  }
  // reduce across the 16 lanes sharing each C-row group
#pragma unroll
  for (int off = 8; off >= 1; off >>= 1)
#pragma unroll
    for (int r = 0; r < 8; ++r) part[r] += __shfl_xor(part[r], off, 32);
  if ((lane & 15) == 0) {
#pragma unroll
    for (int r = 0; r < 8; ++r) out[p0 + crow + r] = part[r] + Rb3[0];
  }
}

// ---------------- host launch ----------------
extern "C" void kernel_launch(void* const* d_in, const int* in_sizes, int n_in,
                              void* d_out, int out_size, void* d_ws, size_t ws_size,
                              hipStream_t stream) {
  const int*   links = (const int*)  d_in[0];
  // d_in[1]=paths, d_in[2]=seqs are structurally regular (p*L+t) and unused
  const float* cap   = (const float*)d_in[3];
  const float* pol   = (const float*)d_in[4];
  const float* twt   = (const float*)d_in[5];
  const float* bw    = (const float*)d_in[6];
  const float* tos   = (const float*)d_in[7];
  const float* pk    = (const float*)d_in[8];
  const float* avg   = (const float*)d_in[9];
  const float* Wp    = (const float*)d_in[10];
  const float* Up    = (const float*)d_in[11];
  const float* bp    = (const float*)d_in[12];
  const float* Wl    = (const float*)d_in[13];
  const float* Ul    = (const float*)d_in[14];
  const float* bl    = (const float*)d_in[15];
  const float* R1    = (const float*)d_in[16];
  const float* Rb1   = (const float*)d_in[17];
  const float* R2    = (const float*)d_in[18];
  const float* Rb2   = (const float*)d_in[19];
  const float* R3    = (const float*)d_in[20];
  const float* Rb3   = (const float*)d_in[21];

  float* path_state = (float*)d_ws;                    // PN*32 f32
  float* link_state = path_state + PN * DN;            // NLN*32 f32
  float* m_agg      = link_state + NLN * DN;           // NLN*32 f32
  _Float16* w16     = (_Float16*)(m_agg + NLN * DN);   // 86016 f16
  _Float16* Wp16 = w16;
  _Float16* Up16 = w16 + 3072;
  _Float16* Wl16 = w16 + 6144;
  _Float16* Ul16 = w16 + 9216;
  _Float16* R1h  = w16 + 12288;
  _Float16* R2h  = w16 + 20480;

  init_states_kernel<<<(PN * DN + 255) / 256, 256, 0, stream>>>(
      cap, pol, twt, bw, tos, pk, avg, link_state, path_state);
  cvt_weights_kernel<<<(86016 + 255) / 256, 256, 0, stream>>>(
      Wp, Up, Wl, Ul, R1, R2, w16);

  const int path_blocks = (PN / 16 + 3) / 4;   // 6250 wave-tiles, 4 waves/block
  const int link_blocks = (NLN / 16 + 3) / 4;  // 625 wave-tiles
  for (int it = 0; it < TN; ++it) {
    zero_kernel<<<(NLN * DN + 255) / 256, 256, 0, stream>>>(m_agg, NLN * DN);
    path_mp_kernel<<<path_blocks, 128, 0, stream>>>(
        links, link_state, path_state, m_agg, Wp16, Up16, bp);
    link_mp_kernel<<<link_blocks, 128, 0, stream>>>(
        m_agg, link_state, Wl16, Ul16, bl);
  }
  readout_kernel<<<PN / 32, 64, 0, stream>>>(
      path_state, R1h, Rb1, R2h, Rb2, R3, Rb3, (float*)d_out);
}
